// InteractionAttention_79087527789123
// MI455X (gfx1250) — matrix operand order repaired and checked by reading, not compiled, not run on hardware
//
#include <hip/hip_runtime.h>
#include <hip/hip_bf16.h>
#include <math.h>

// ---------------------------------------------------------------------------
// InteractionAttention fused kernel for MI455X (gfx1250, wave32, WMMA)
// B=4 L=100 N=48 D=256 H=4 C=4 EXP=4
// ---------------------------------------------------------------------------

typedef __attribute__((ext_vector_type(16))) _Float16 v16h;
typedef __attribute__((ext_vector_type(8)))  _Float16 v8h;
typedef __attribute__((ext_vector_type(8)))  float    v8f;

static constexpr int Bb = 4, Ll = 100, Nn = 48, Dd = 256, Hh = 4;

// ---- WMMA fragment helpers (contiguous k order, not the ISA table order; A and B match) ------------------
// A (16x32 f16, M x K): lane m=lane&15 holds row m; element e -> K = e + hi*16
// (hi = lane>>4, contiguous, matching frag_bt). Two 16B loads.
__device__ __forceinline__ v16h frag_a(const _Float16* base, int strideH,
                                       int row, int k0, int hi) {
  const v8h* p0 = (const v8h*)(base + row * strideH + k0 + hi * 16);
  const v8h* p1 = (const v8h*)(base + row * strideH + k0 + hi * 16 + 8);
  v8h lo = *p0, hv = *p1;
  return __builtin_shufflevector(lo, hv, 0,1,2,3,4,5,6,7,8,9,10,11,12,13,14,15);
}

// B (32x16 f16, K x N) loaded from N-major ("Bt") storage: lane n=lane&15 holds
// column n; lanes 0-15 cover K=k..k+15, lanes 16-31 K=k+16..k+31 (contiguous).
__device__ __forceinline__ v16h frag_bt(const _Float16* base, int strideH,
                                        int row, int k0) {
  const v8h* p = (const v8h*)(base + row * strideH + k0);
  v8h lo = p[0], hv = p[1];
  return __builtin_shufflevector(lo, hv, 0,1,2,3,4,5,6,7,8,9,10,11,12,13,14,15);
}

__device__ __forceinline__ v8f wmma_f(v16h a, v16h b, v8f c) {
  return __builtin_amdgcn_wmma_f32_16x16x32_f16(false, a, false, b,
                                                (short)0, c, false, false);
}

// ---------------------------------------------------------------------------
// Prep kernel 1: per-batch LayerNorm over (rows, cols) jointly, f32 -> f16.
// Optional per-row mask multiply (embedding_mask) folded into the output.
// ---------------------------------------------------------------------------
__global__ __launch_bounds__(256)
void ia_ln_batch(const float* __restrict__ x, const float* __restrict__ mask,
                 _Float16* __restrict__ o, int rows, int cols) {
  const int b = blockIdx.x;
  const int cnt = rows * cols;
  const float* xb = x + b * cnt;
  __shared__ float rs[256], rq[256];
  float s = 0.f, q = 0.f;
  for (int i = threadIdx.x; i < cnt; i += 256) {
    float v = xb[i]; s += v; q += v * v;
  }
  rs[threadIdx.x] = s; rq[threadIdx.x] = q;
  __syncthreads();
  for (int off = 128; off > 0; off >>= 1) {
    if (threadIdx.x < off) {
      rs[threadIdx.x] += rs[threadIdx.x + off];
      rq[threadIdx.x] += rq[threadIdx.x + off];
    }
    __syncthreads();
  }
  const float m = rs[0] / (float)cnt;
  const float var = rq[0] / (float)cnt - m * m;
  const float r = rsqrtf(var + 1e-5f);
  for (int i = threadIdx.x; i < cnt; i += 256) {
    float v = (xb[i] - m) * r;
    if (mask) v *= mask[b * rows + i / cols];
    o[b * cnt + i] = (_Float16)v;
  }
}

// ---------------------------------------------------------------------------
// Prep kernel 2: head-deinterleave Wq/Wk/Wv/Wg (1024,256) -> f16 [h][dout][din]
// ---------------------------------------------------------------------------
__global__ __launch_bounds__(256)
void ia_conv_head(const float* __restrict__ w, _Float16* __restrict__ o) {
  const int t = blockIdx.x * 256 + threadIdx.x;        // 0..262143
  const int h = t >> 16, rem = t & 65535;
  const int dout = rem >> 8, din = rem & 255;
  o[t] = (_Float16)w[(dout * 4 + h) * 256 + din];
}

// Wo (256,1024) -> f16 [h][e][d] (e = out feat, d = per-head hidden)
__global__ __launch_bounds__(256)
void ia_conv_wo(const float* __restrict__ w, _Float16* __restrict__ o) {
  const int t = blockIdx.x * 256 + threadIdx.x;
  const int h = t >> 16, e = (t >> 8) & 255, d = t & 255;
  o[t] = (_Float16)w[e * 1024 + d * 4 + h];
}

// plain f32 -> f16 copy (W1, W2 already (out,in) row-major == Bt layout)
__global__ __launch_bounds__(256)
void ia_conv_copy(const float* __restrict__ w, _Float16* __restrict__ o, int n) {
  const int t = blockIdx.x * 256 + threadIdx.x;
  if (t < n) o[t] = (_Float16)w[t];
}

// bias[b,i,j,h] = sum_c mm[b,i,j,c] * Wm[h,c]   (C=4, H=4)
__global__ __launch_bounds__(256)
void ia_bias(const float* __restrict__ mm, const float* __restrict__ wm,
             float* __restrict__ o) {
  const int t = blockIdx.x * 256 + threadIdx.x;
  if (t < Bb * Nn * Nn) {
    float c0 = mm[t*4+0], c1 = mm[t*4+1], c2 = mm[t*4+2], c3 = mm[t*4+3];
#pragma unroll
    for (int h = 0; h < Hh; ++h)
      o[t*4 + h] = c0*wm[h*4+0] + c1*wm[h*4+1] + c2*wm[h*4+2] + c3*wm[h*4+3];
  }
}

// ---------------------------------------------------------------------------
// Fused main kernel: one workgroup per (b,l); 8 wave32 waves; ~197 KB LDS.
// ---------------------------------------------------------------------------
__global__ __launch_bounds__(256)
void ia_fused(const _Float16* __restrict__ erh, const _Float16* __restrict__ emh,
              const _Float16* __restrict__ wq4, const _Float16* __restrict__ wk4,
              const _Float16* __restrict__ wv4, const _Float16* __restrict__ wg4,
              const _Float16* __restrict__ wo4, const _Float16* __restrict__ w1c,
              const _Float16* __restrict__ w2c, const float* __restrict__ biasf,
              const unsigned char* __restrict__ mmask,
              const float* __restrict__ bo, const float* __restrict__ lnw,
              const float* __restrict__ lnb, const float* __restrict__ b1,
              const float* __restrict__ b2, float* __restrict__ out) {
  extern __shared__ char smem[];
  _Float16* aa   = (_Float16*)smem;                 // 48x256 f16  (24576 B)
  _Float16* qs   = aa + Nn * Dd;                    // 48x256 f16
  _Float16* ks   = qs + Nn * Dd;                    // 48x256 f16
  _Float16* vT   = ks + Nn * Dd;                    // 256x64 f16 (j padded)
  float*    attf = (float*)(vT + Dd * 64);          // 48x48 f32
  _Float16* atth = (_Float16*)(attf + Nn * Nn);     // 48x64 f16 (j padded)
  _Float16* og   = atth + Nn * 64;                  // 48x256 f16
  float*    y    = (float*)(og + Nn * Dd);          // 48x256 f32 accumulator
  _Float16* ub   = (_Float16*)(y + Nn * Dd);        // 48x64 f16 MLP chunk

  const int tid = threadIdx.x;
  const int wv  = tid >> 5;        // wave id 0..7
  const int lane = tid & 31;
  const int hi  = lane >> 4;       // half-wave select
  const int nl  = lane & 15;       // lane-in-16

  const int bl = blockIdx.x;
  const int b = bl / Ll, l = bl % Ll;

  // ---- stage 0: aa[i][d] = er_norm[l][d] * em_norm[i][d] (mask folded in em)
  const _Float16* erow = erh + (b * Ll + l) * Dd;
  const _Float16* emB  = emh + b * Nn * Dd;
  for (int idx = tid; idx < Nn * Dd; idx += 256)
    aa[idx] = erow[idx & 255] * emB[idx];
  for (int idx = tid; idx < Dd * 16; idx += 256)        // zero vT K-padding
    vT[(idx >> 4) * 64 + 48 + (idx & 15)] = (_Float16)0.f;
  __syncthreads();

  for (int h = 0; h < Hh; ++h) {
    const _Float16* WQ = wq4 + h * 65536;
    const _Float16* WK = wk4 + h * 65536;
    const _Float16* WV = wv4 + h * 65536;
    const _Float16* WG = wg4 + h * 65536;
    const _Float16* WO = wo4 + h * 65536;

    // ---- stage 1: q = aa@Wqh^T, k = aa@Wkh^T, v = aa@Wvh^T (v transposed)
    for (int tt = wv; tt < 48; tt += 8) {
      const int mb = tt % 3, nb = tt / 3;
      v8f cq = {}, ck = {}, cv = {};
#pragma unroll
      for (int kb = 0; kb < 8; ++kb) {
        v16h a = frag_a(aa, Dd, mb * 16 + nl, kb * 32, hi);
        const int brow = nb * 16 + nl, bcol = kb * 32 + hi * 16;
        cq = wmma_f(a, frag_bt(WQ, Dd, brow, bcol), cq);
        ck = wmma_f(a, frag_bt(WK, Dd, brow, bcol), ck);
        cv = wmma_f(a, frag_bt(WV, Dd, brow, bcol), cv);
      }
#pragma unroll
      for (int g = 0; g < 8; ++g) {
        const int row = mb * 16 + g + hi * 8, col = nb * 16 + nl;
        qs[row * Dd + col] = (_Float16)cq[g];
        ks[row * Dd + col] = (_Float16)ck[g];
        vT[col * 64 + row] = (_Float16)cv[g];       // d-major for att@v B-op
      }
    }
    __syncthreads();

    // ---- stage 2: att = q @ k^T + bias, mask, scale
    for (int tt = wv; tt < 9; tt += 8) {
      const int mb = tt / 3, nb = tt % 3;
      v8f c = {};
#pragma unroll
      for (int kb = 0; kb < 8; ++kb) {
        v16h a  = frag_a(qs, Dd, mb * 16 + nl, kb * 32, hi);
        v16h bb = frag_bt(ks, Dd, nb * 16 + nl, kb * 32 + hi * 16);
        c = wmma_f(a, bb, c);
      }
#pragma unroll
      for (int g = 0; g < 8; ++g) {
        const int i = mb * 16 + g + hi * 8, j = nb * 16 + nl;
        const int ij = (b * Nn + i) * Nn + j;
        attf[i * Nn + j] = mmask[ij] ? (c[g] + biasf[ij * 4 + h]) * 0.0625f
                                     : -3.0e38f;
      }
    }
    __syncthreads();

    // ---- stage 3: row softmax -> f16 (with K padding zeros)
    if (tid < Nn) {
      const int r = tid;
      float mx = -3.4e38f;
      for (int j = 0; j < Nn; ++j) mx = fmaxf(mx, attf[r * Nn + j]);
      float s = 0.f;
      for (int j = 0; j < Nn; ++j) s += expf(attf[r * Nn + j] - mx);
      const float inv = 1.f / s;
      for (int j = 0; j < Nn; ++j)
        atth[r * 64 + j] = (_Float16)(expf(attf[r * Nn + j] - mx) * inv);
      for (int j = Nn; j < 64; ++j) atth[r * 64 + j] = (_Float16)0.f;
    }
    __syncthreads();

    // ---- stage 4: out_h = att @ v ; gate = sigmoid(aa@Wgh^T); og = out*gate
    for (int tt = wv; tt < 48; tt += 8) {
      const int mb = tt % 3, nb = tt / 3;
      v8f co = {};
#pragma unroll
      for (int kb = 0; kb < 2; ++kb) {            // K=48 padded to 64
        v16h a  = frag_a(atth, 64, mb * 16 + nl, kb * 32, hi);
        v16h bb = frag_bt(vT, 64, nb * 16 + nl, kb * 32 + hi * 16);
        co = wmma_f(a, bb, co);
      }
      v8f cg = {};
#pragma unroll
      for (int kb = 0; kb < 8; ++kb) {
        v16h a  = frag_a(aa, Dd, mb * 16 + nl, kb * 32, hi);
        v16h bb = frag_bt(WG, Dd, nb * 16 + nl, kb * 32 + hi * 16);
        cg = wmma_f(a, bb, cg);
      }
#pragma unroll
      for (int g = 0; g < 8; ++g) {
        const int row = mb * 16 + g + hi * 8, col = nb * 16 + nl;
        const float gate = 1.f / (1.f + expf(-cg[g]));
        og[row * Dd + col] = (_Float16)(co[g] * gate);
      }
    }
    __syncthreads();

    // ---- stage 5: y += og @ Woh^T   (f32 accumulation across heads)
    for (int tt = wv; tt < 48; tt += 8) {
      const int mb = tt % 3, nb = tt / 3;
      v8f c = {};
      if (h != 0) {
#pragma unroll
        for (int g = 0; g < 8; ++g)
          c[g] = y[(mb * 16 + g + hi * 8) * Dd + nb * 16 + nl];
      }
#pragma unroll
      for (int kb = 0; kb < 8; ++kb) {
        v16h a  = frag_a(og, Dd, mb * 16 + nl, kb * 32, hi);
        v16h bb = frag_bt(WO, Dd, nb * 16 + nl, kb * 32 + hi * 16);
        c = wmma_f(a, bb, c);
      }
#pragma unroll
      for (int g = 0; g < 8; ++g)
        y[(mb * 16 + g + hi * 8) * Dd + nb * 16 + nl] = c[g];
    }
    __syncthreads();
  }

  // ---- stage 6: t = LN(y + bo) * ln_w + ln_b -> aa (f16), per row
  if (tid < Nn) {
    const int r = tid;
    float s = 0.f, q = 0.f;
    for (int c = 0; c < Dd; ++c) {
      float v = y[r * Dd + c] + bo[c]; s += v; q += v * v;
    }
    const float m = s / (float)Dd;
    const float var = q / (float)Dd - m * m;
    const float rs = rsqrtf(var + 1e-5f);
    for (int c = 0; c < Dd; ++c) {
      float v = y[r * Dd + c] + bo[c];
      aa[r * Dd + c] = (_Float16)(((v - m) * rs) * lnw[c] + lnb[c]);
    }
  }
  __syncthreads();

  // ---- stage 7: z = relu(t@W1^T + b1) @ W2^T, chunked over hidden (16x64)
  for (int cb = 0; cb < 16; ++cb) {
    // u chunk: (48 x 64)
    for (int tt = wv; tt < 12; tt += 8) {
      const int mb = tt % 3, cc = tt / 3;
      v8f c = {};
#pragma unroll
      for (int kb = 0; kb < 8; ++kb) {
        v16h a  = frag_a(aa, Dd, mb * 16 + nl, kb * 32, hi);
        v16h bb = frag_bt(w1c, Dd, cb * 64 + cc * 16 + nl, kb * 32 + hi * 16);
        c = wmma_f(a, bb, c);
      }
#pragma unroll
      for (int g = 0; g < 8; ++g) {
        const int row = mb * 16 + g + hi * 8, col = cc * 16 + nl;
        ub[row * 64 + col] = (_Float16)fmaxf(c[g] + b1[cb * 64 + col], 0.f);
      }
    }
    __syncthreads();
    // z accumulate: (48 x 256) += u_chunk @ W2_chunk^T
    for (int tt = wv; tt < 48; tt += 8) {
      const int mb = tt % 3, nb = tt / 3;
      v8f c = {};
      if (cb != 0) {
#pragma unroll
        for (int g = 0; g < 8; ++g)
          c[g] = y[(mb * 16 + g + hi * 8) * Dd + nb * 16 + nl];
      }
#pragma unroll
      for (int kb = 0; kb < 2; ++kb) {
        v16h a  = frag_a(ub, 64, mb * 16 + nl, kb * 32, hi);
        v16h bb = frag_bt(w2c, 1024, nb * 16 + nl, cb * 64 + kb * 32 + hi * 16);
        c = wmma_f(a, bb, c);
      }
#pragma unroll
      for (int g = 0; g < 8; ++g)
        y[(mb * 16 + g + hi * 8) * Dd + nb * 16 + nl] = c[g];
    }
    __syncthreads();
  }

  // ---- stage 8: out[b,l,e] = mean_i z[i,e] + b2[e]
  if (tid < Dd) {
    float s = 0.f;
    for (int i = 0; i < Nn; ++i) s += y[i * Dd + tid];
    out[(b * Ll + l) * Dd + tid] = s * (1.f / (float)Nn) + b2[tid];
  }
}

// ---------------------------------------------------------------------------
// Host launch
// ---------------------------------------------------------------------------
extern "C" void kernel_launch(void* const* d_in, const int* in_sizes, int n_in,
                              void* d_out, int out_size, void* d_ws, size_t ws_size,
                              hipStream_t stream) {
  const float* energy = (const float*)d_in[0];
  const float* embed  = (const float*)d_in[1];
  const float* mm     = (const float*)d_in[2];
  const float* emask  = (const float*)d_in[3];
  const unsigned char* mmask = (const unsigned char*)d_in[4];   // bool mask
  const float* Wm = (const float*)d_in[5];
  const float* Wq = (const float*)d_in[6];
  const float* Wk = (const float*)d_in[7];
  const float* Wv = (const float*)d_in[8];
  const float* Wg = (const float*)d_in[9];
  const float* Wo = (const float*)d_in[10];
  const float* bo = (const float*)d_in[11];
  const float* lnw = (const float*)d_in[12];
  const float* lnb = (const float*)d_in[13];
  const float* W1 = (const float*)d_in[14];
  const float* b1 = (const float*)d_in[15];
  const float* W2 = (const float*)d_in[16];
  const float* b2 = (const float*)d_in[17];

  char* ws = (char*)d_ws;
  _Float16* erh  = (_Float16*)(ws + 0);         // B*L*D f16      204800
  _Float16* emh  = (_Float16*)(ws + 204800);    // B*N*D f16       98304
  _Float16* wq4  = (_Float16*)(ws + 303104);    // [H][256][256]  524288
  _Float16* wk4  = (_Float16*)(ws + 827392);
  _Float16* wv4  = (_Float16*)(ws + 1351680);
  _Float16* wg4  = (_Float16*)(ws + 1875968);
  _Float16* wo4  = (_Float16*)(ws + 2400256);
  _Float16* w1c  = (_Float16*)(ws + 2924544);   // [1024][256]    524288
  _Float16* w2c  = (_Float16*)(ws + 3448832);   // [256][1024]    524288
  float*    biasf= (float*)   (ws + 3973120);   // B*N*N*H f32    147456

  // prep: per-batch layernorms (mask folded into embedding)
  ia_ln_batch<<<Bb, 256, 0, stream>>>(energy, nullptr, erh, Ll, Dd);
  ia_ln_batch<<<Bb, 256, 0, stream>>>(embed,  emask,   emh, Nn, Dd);
  // prep: weight conversion to f16 Bt layouts (L2-resident afterwards)
  ia_conv_head<<<1024, 256, 0, stream>>>(Wq, wq4);
  ia_conv_head<<<1024, 256, 0, stream>>>(Wk, wk4);
  ia_conv_head<<<1024, 256, 0, stream>>>(Wv, wv4);
  ia_conv_head<<<1024, 256, 0, stream>>>(Wg, wg4);
  ia_conv_wo  <<<1024, 256, 0, stream>>>(Wo, wo4);
  ia_conv_copy<<<1024, 256, 0, stream>>>(W1, w1c, 262144);
  ia_conv_copy<<<1024, 256, 0, stream>>>(W2, w2c, 262144);
  ia_bias<<<36, 256, 0, stream>>>(mm, Wm, biasf);

  // fused attention + transition: one workgroup per (b,l)
  const size_t smem = 201728;   // fits 320KB CDNA5 WGP LDS
  ia_fused<<<Bb * Ll, 256, smem, stream>>>(erh, emh, wq4, wk4, wv4, wg4, wo4,
                                           w1c, w2c, biasf, mmask, bo, lnw, lnb,
                                           b1, b2, (float*)d_out);
}
